// SimpleRNN_48352741818559
// MI455X (gfx1250) — matrix operand order repaired and checked by reading, not compiled, not run on hardware
//
#include <hip/hip_runtime.h>

// ---------------------------------------------------------------------------
// SimpleRNN fused for gfx1250 (MI455X): bf16 WMMA, batch-parallel recurrence.
//   B=256, S=512, I=128, H=512, O=128
// ---------------------------------------------------------------------------

typedef __attribute__((ext_vector_type(16))) __bf16        v16bf;
typedef __attribute__((ext_vector_type(4)))  __bf16        v4bf;
typedef __attribute__((ext_vector_type(8)))  float         v8f;
typedef __attribute__((ext_vector_type(4)))  float         v4f;
typedef __attribute__((ext_vector_type(4)))  unsigned int  u32x4;

// TDM descriptor vector types (probe-confirmed builtin signatures)
typedef __attribute__((ext_vector_type(4))) unsigned int tdm_u32x4;
typedef __attribute__((ext_vector_type(8))) int          tdm_i32x8;
typedef __attribute__((ext_vector_type(4))) int          tdm_i32x4;

#define B_ 256
#define S_ 512
#define I_ 128
#define H_ 512
#define O_ 128

// -------------------------- fp32 -> bf16 convert ---------------------------
__global__ void cvt_f32_bf16(const float* __restrict__ src,
                             __bf16* __restrict__ dst, int n4) {
    int i      = blockIdx.x * blockDim.x + threadIdx.x;
    int stride = gridDim.x * blockDim.x;
    for (; i < n4; i += stride) {
        v4f f = *(const v4f*)(src + 4 * (size_t)i);
        v4bf o;
        o.x = (__bf16)f.x; o.y = (__bf16)f.y;
        o.z = (__bf16)f.z; o.w = (__bf16)f.w;
        *(v4bf*)(dst + 4 * (size_t)i) = o;
    }
}

// ------------------------------ helpers ------------------------------------
__device__ __forceinline__ v8f wmma_bf16(v16bf a, v16bf b, v8f c) {
    // (neg_a, A, neg_b, B, c_mod, C, reuse_a, reuse_b)
    return __builtin_amdgcn_wmma_f32_16x16x32_bf16(false, a, false, b,
                                                   (short)0, c, false, false);
}

// Fast tanh: tanh(x) = 1 - 2/(exp(2x)+1), exp via v_exp_f32 (exp2).
// Saturates correctly at +/-inf; precision far exceeds bf16 storage of h.
__device__ __forceinline__ float fast_tanh(float x) {
    float e = __builtin_amdgcn_exp2f(x * 2.8853900817779268f); // exp(2x)
    return 1.0f - 2.0f * __builtin_amdgcn_rcpf(e + 1.0f);
}

// A fragment (16x32, bf16), base points at (row m0, col k0), row stride ldk.
// ISA layout: lanes 0-15 hold K 0..7 (v0..3) and 16..23 (v4..7); lanes 16-31
// hold K 8..15 and 24..31.  => two 16-byte chunks per lane at +hf*8 and +16.
__device__ __forceinline__ v16bf load_frag_a(const __bf16* base, int ldk,
                                             int lane) {
    int r = lane & 15, hf = lane >> 4;
    const __bf16* p = base + (size_t)r * ldk + hf * 8;
    union { v16bf v; u32x4 q[2]; } u;
    u.q[0] = *(const u32x4*)(const void*)(p);
    u.q[1] = *(const u32x4*)(const void*)(p + 16);
    return u.v;
}

// B fragment (32x16, bf16) from row-major W[n][k] (so B[k][n] = W[n][k]).
// Lane L holds W[n0 + (L&15)][k0 + (L>>4)*8 .. +7] and [+16 .. +23].
__device__ __forceinline__ v16bf load_frag_b(const __bf16* Wnk, int ldk,
                                             int n0, int k0, int lane) {
    int r = lane & 15, hf = lane >> 4;
    const __bf16* p = Wnk + (size_t)(n0 + r) * ldk + k0 + hf * 8;
    union { v16bf v; u32x4 q[2]; } u;
    u.q[0] = *(const u32x4*)(const void*)(p);
    u.q[1] = *(const u32x4*)(const void*)(p + 16);
    return u.v;
}

// ------------------------------ fused RNN ----------------------------------
// Grid: 16 blocks (one 16-row batch tile each). Block: 256 threads = 8 waves.
// Wave w owns output columns [w*64, w*64+64) of H during the scan.
__global__ void __launch_bounds__(256) rnn_fused(
    const __bf16* __restrict__ xb,    // [B,S,I] bf16
    const __bf16* __restrict__ wih,   // [H,I]   bf16
    const __bf16* __restrict__ whh,   // [H,H]   bf16
    const __bf16* __restrict__ wfc,   // [O,H]   bf16
    const float*  __restrict__ bih,
    const float*  __restrict__ bhh,
    const float*  __restrict__ bfc,
    float*        __restrict__ out)   // [B,O] f32
{
    extern __shared__ char smem[];
    __bf16* wih_s = (__bf16*)smem;                          // 128 KB
    __bf16* hbuf0 = (__bf16*)(smem + (size_t)H_ * I_ * 2);  // 16 KB
    __bf16* hbuf1 = hbuf0 + 16 * H_;                        // 16 KB

    const int tid  = threadIdx.x;
    const int lane = tid & 31;
    const int wave = tid >> 5;          // 0..7
    const int m0   = blockIdx.x * 16;   // batch-row tile

    // ---- Stage W_ih into LDS via the Tensor Data Mover (one 2D descriptor:
    //      512 rows x 128 bf16, contiguous), issued by wave 0. ----
#if __has_builtin(__builtin_amdgcn_tensor_load_to_lds)
    if (wave == 0) {
        unsigned long long ga = (unsigned long long)(const void*)wih;
        unsigned lds_off = (unsigned)(unsigned long long)(void*)wih_s;
        tdm_u32x4 g0;
        g0.x = 0x1u;                                   // count=1, user mode
        g0.y = lds_off;                                // lds_addr (bytes)
        g0.z = (unsigned)(ga & 0xFFFFFFFFu);           // global_addr[31:0]
        g0.w = (unsigned)((ga >> 32) & 0x01FFFFFFu)    // global_addr[56:32]
             | (2u << 30);                             // type = 2 (image)
        tdm_i32x8 g1;
        g1[0] = (int)(1u << 16);            // data_size = 1 -> 2 bytes
        g1[1] = (int)((unsigned)I_ << 16);  // tensor_dim0[15:0] = 128
        g1[2] = (int)((unsigned)H_ << 16);  // tensor_dim1[15:0] = 512
        g1[3] = (int)((unsigned)I_ << 16);  // tile_dim0 = 128
        g1[4] = (int)H_;                    // tile_dim1 = 512
        g1[5] = (int)I_;                    // tensor_dim0_stride = 128
        g1[6] = 0;
        g1[7] = 0;
        tdm_i32x4 gz; gz[0] = 0; gz[1] = 0; gz[2] = 0; gz[3] = 0;
#if defined(__clang_major__) && (__clang_major__ >= 23)
        tdm_i32x8 gz8;
        gz8[0]=0; gz8[1]=0; gz8[2]=0; gz8[3]=0;
        gz8[4]=0; gz8[5]=0; gz8[6]=0; gz8[7]=0;
        __builtin_amdgcn_tensor_load_to_lds(g0, g1, gz, gz, gz8, 0);
#else
        __builtin_amdgcn_tensor_load_to_lds(g0, g1, gz, gz, 0);
#endif
        __builtin_amdgcn_s_wait_tensorcnt(0);
    }
#else
    for (int i = tid * 4; i < H_ * I_; i += 256 * 4)
        *(v4bf*)(wih_s + i) = *(const v4bf*)(wih + i);
#endif

    // h0 = 0 (both double buffers, contiguous).
    v4bf z; z.x = (__bf16)0.0f; z.y = (__bf16)0.0f;
    z.z = (__bf16)0.0f; z.w = (__bf16)0.0f;
    for (int i = tid * 4; i < 2 * 16 * H_; i += 256 * 4)
        *(v4bf*)(hbuf0 + i) = z;
    __syncthreads();

    v8f zero8;
    #pragma unroll
    for (int i = 0; i < 8; ++i) zero8[i] = 0.0f;

    const int nlo = lane & 15;          // N within tile
    const int mb  = (lane >> 4) * 8;    // M base for C/D rows
    const size_t xrowstride = (size_t)S_ * I_;

    // Hoist per-thread fused bias (loop-invariant over t).
    float bv[4];
    #pragma unroll
    for (int j = 0; j < 4; ++j) {
        int nc = wave * 64 + j * 16 + nlo;
        bv[j] = bih[nc] + bhh[nc];
    }

    for (int t = 0; t < S_; ++t) {
        const __bf16* hc = (t & 1) ? hbuf1 : hbuf0;
        __bf16*       hn = (t & 1) ? hbuf0 : hbuf1;

        v8f acc[4];
        #pragma unroll
        for (int j = 0; j < 4; ++j) acc[j] = zero8;

        // ---- input projection: K = I (128), W_ih from LDS ----
        const __bf16* xbase = xb + ((size_t)m0 * S_ + t) * I_;
        #pragma unroll
        for (int k0 = 0; k0 < I_; k0 += 32) {
            v16bf a = load_frag_a(xbase + k0, (int)xrowstride, lane);
            #pragma unroll
            for (int j = 0; j < 4; ++j)
                acc[j] = wmma_bf16(a,
                    load_frag_b(wih_s, I_, wave * 64 + j * 16, k0, lane),
                    acc[j]);
        }

        // ---- recurrent projection: K = H (512), W_hh streams from L2 ----
        #pragma unroll 4
        for (int k0 = 0; k0 < H_; k0 += 32) {
            __builtin_prefetch(whh + (size_t)(wave * 64 + nlo) * H_ + k0 + 128,
                               0, 0);
            v16bf a = load_frag_a(hc + k0, H_, lane);
            #pragma unroll
            for (int j = 0; j < 4; ++j)
                acc[j] = wmma_bf16(a,
                    load_frag_b(whh, H_, wave * 64 + j * 16, k0, lane),
                    acc[j]);
        }

        // ---- bias + tanh, write next-h to LDS (bf16) ----
        #pragma unroll
        for (int j = 0; j < 4; ++j) {
            int nc = wave * 64 + j * 16 + nlo;
            #pragma unroll
            for (int i = 0; i < 8; ++i) {
                float v = fast_tanh(acc[j][i] + bv[j]);
                hn[(size_t)(mb + i) * H_ + nc] = (__bf16)v;
            }
        }
        __syncthreads();
    }

    // ---- final FC: out[16,128] = h_final @ W_fc^T + b_fc  (S even -> hbuf0)
    {
        const __bf16* hf = hbuf0;
        v8f acc = zero8;
        #pragma unroll 4
        for (int k0 = 0; k0 < H_; k0 += 32) {
            v16bf a = load_frag_a(hf + k0, H_, lane);
            acc = wmma_bf16(a, load_frag_b(wfc, H_, wave * 16, k0, lane), acc);
        }
        int nc = wave * 16 + nlo;
        float bvf = bfc[nc];
        #pragma unroll
        for (int i = 0; i < 8; ++i)
            out[(size_t)(m0 + mb + i) * O_ + nc] = acc[i] + bvf;
    }
}

// ------------------------------ launcher -----------------------------------
extern "C" void kernel_launch(void* const* d_in, const int* in_sizes, int n_in,
                              void* d_out, int out_size, void* d_ws,
                              size_t ws_size, hipStream_t stream) {
    (void)in_sizes; (void)n_in; (void)out_size; (void)ws_size;
    const float* x    = (const float*)d_in[0];
    const float* W_ih = (const float*)d_in[1];
    const float* W_hh = (const float*)d_in[2];
    const float* b_ih = (const float*)d_in[3];
    const float* b_hh = (const float*)d_in[4];
    const float* W_fc = (const float*)d_in[5];
    const float* b_fc = (const float*)d_in[6];
    float* out = (float*)d_out;

    // Workspace layout (bf16 copies): x | W_ih | W_hh | W_fc  (~34.3 MB)
    __bf16* xb  = (__bf16*)d_ws;
    __bf16* wih = xb  + (size_t)B_ * S_ * I_;
    __bf16* whh = wih + (size_t)H_ * I_;
    __bf16* wfc = whh + (size_t)H_ * H_;

    cvt_f32_bf16<<<4096, 256, 0, stream>>>(x,    xb,  (B_ * S_ * I_) / 4);
    cvt_f32_bf16<<<64,   256, 0, stream>>>(W_ih, wih, (H_ * I_) / 4);
    cvt_f32_bf16<<<256,  256, 0, stream>>>(W_hh, whh, (H_ * H_) / 4);
    cvt_f32_bf16<<<64,   256, 0, stream>>>(W_fc, wfc, (O_ * H_) / 4);

    size_t shmem = (size_t)H_ * I_ * 2      // W_ih in LDS: 128 KB
                 + (size_t)2 * 16 * H_ * 2; // h double buffer: 32 KB
    rnn_fused<<<B_ / 16, 256, shmem, stream>>>(xb, wih, whh, wfc,
                                               b_ih, b_hh, b_fc, out);
}
